// CrossAttention_67800353735345
// MI455X (gfx1250) — compile-verified, matrix operand order checked by reading, not run on hardware
//
#include <hip/hip_runtime.h>
#include <hip/hip_bf16.h>
#include <cstddef>

typedef __attribute__((ext_vector_type(16))) _Float16 v16h;
typedef __attribute__((ext_vector_type(8)))  _Float16 v8h;
typedef __attribute__((ext_vector_type(4)))  _Float16 v4h;
typedef __attribute__((ext_vector_type(8)))  float    v8f;
typedef __attribute__((ext_vector_type(4)))  int      v4i;

#define B_SZ 8
#define NQ_  1024
#define NK_  1024
#define DQ_  1024
#define H_   16
#define HD_  64
#define SCALE_ 0.125f   // 64^-0.5
#define LDT 40          // GEMM LDS row stride (f16), 80B = 16B aligned
#define LDL 72          // logits LDS row stride (64 + pad), 144B = 16B aligned

#define AS1 __attribute__((address_space(1)))
#define AS3 __attribute__((address_space(3)))

#if defined(__AMDGCN__) && __has_builtin(__builtin_amdgcn_global_load_async_to_lds_b128)
#define HAVE_ASYNC_LDS 1
#else
#define HAVE_ASYNC_LDS 0
#endif

// 16B global -> LDS copy. Async (ASYNCcnt-tracked, no VGPR round-trip) on
// CDNA5 when the builtin is available; plain copy otherwise.
__device__ inline void cp_async_b128(void* dst_lds, const void* src_g) {
#if HAVE_ASYNC_LDS
    __builtin_amdgcn_global_load_async_to_lds_b128(
        (AS1 v4i*)src_g, (AS3 v4i*)dst_lds, 0, 0);
#else
    *(v4i*)dst_lds = *(const v4i*)src_g;
#endif
}

__device__ inline void async_wait() {
#if HAVE_ASYNC_LDS
#if __has_builtin(__builtin_amdgcn_s_wait_asynccnt)
    __builtin_amdgcn_s_wait_asynccnt(0);
#else
    asm volatile("s_wait_asynccnt 0x0" ::: "memory");
#endif
#endif
}

union FragA { v16h v; v8h h[2]; };

// Load a 16x32 f16 fragment from LDS laid out row-major with row stride LD.
// CDNA5 16-bit A/B layout: lanes 0-15 = rows, k = hi*8..+7 in VGPR0-3 and
// k+16 in VGPR4-7 (hi = lane>>4) -> two ds_load_b128 per fragment.
template<int LD>
__device__ inline v16h load_frag(const _Float16* base, int lane) {
    FragA f;
    const int r  = lane & 15;
    const int hk = (lane >> 4) << 3;
    f.h[0] = *(const v8h*)(base + r * LD + hk);
    f.h[1] = *(const v8h*)(base + r * LD + 16 + hk);
    return f.v;
}

__device__ inline v8f wmma_f16(v16h a, v16h b, v8f c) {
    return __builtin_amdgcn_wmma_f32_16x16x32_f16(false, a, false, b, (short)0, c, false, false);
}

// ---------------------------------------------------------------------------
// Generic GEMM + bias: out[M,N] = X[M,K] @ W[K,N] + bias. X fp32 or f16,
// W/bias fp32, out fp32 or f16. Block tile 128(M) x 64(N), 8 waves of 32x32.
// Double-buffered LDS: one barrier per K-step.
// ---------------------------------------------------------------------------
template<typename TX, typename TO>
__global__ __launch_bounds__(256)
void gemm_bias_kernel(const TX* __restrict__ X, const float* __restrict__ W,
                      const float* __restrict__ bias, TO* __restrict__ out,
                      int M, int N, int K)
{
    __shared__ _Float16 sA[2][128 * LDT];
    __shared__ _Float16 sB[2][64 * LDT];
    const int tid = threadIdx.x, lane = tid & 31, wave = tid >> 5;
    const int wm = (wave >> 1) << 5, wn = (wave & 1) << 5;
    const int m0 = blockIdx.y * 128, n0 = blockIdx.x * 64;

    auto stage = [&](int buf, int k0) {
        // ---- A tile: 128 rows x 32 k ----
        if constexpr (sizeof(TX) == 4) {
            #pragma unroll
            for (int t = 0; t < 4; ++t) {
                const int e = (tid + t * 256) << 2;
                const int r = e >> 5, c = e & 31;
                const float4 v = *(const float4*)(X + (size_t)(m0 + r) * K + k0 + c);
                v4h h = { (_Float16)v.x, (_Float16)v.y, (_Float16)v.z, (_Float16)v.w };
                *(v4h*)(&sA[buf][r * LDT + c]) = h;
            }
        } else {
            #pragma unroll
            for (int t = 0; t < 2; ++t) {
                const int e = (tid + t * 256) << 3;
                const int r = e >> 5, c = e & 31;
                cp_async_b128(&sA[buf][r * LDT + c],
                              (const _Float16*)X + (size_t)(m0 + r) * K + k0 + c);
            }
        }
        // ---- B tile transposed: sB[n][k] = W[k0+k][n0+n], fp32 -> f16 ----
        #pragma unroll
        for (int t = 0; t < 2; ++t) {
            const int e  = (tid + t * 256) << 2;
            const int kk = e >> 6, n = e & 63;
            const float4 v = *(const float4*)(W + (size_t)(k0 + kk) * N + n0 + n);
            sB[buf][(n + 0) * LDT + kk] = (_Float16)v.x;
            sB[buf][(n + 1) * LDT + kk] = (_Float16)v.y;
            sB[buf][(n + 2) * LDT + kk] = (_Float16)v.z;
            sB[buf][(n + 3) * LDT + kk] = (_Float16)v.w;
        }
    };

    v8f acc00 = {}, acc01 = {}, acc10 = {}, acc11 = {};
    stage(0, 0);
    for (int k0 = 0; k0 < K; k0 += 32) {
        const int buf = (k0 >> 5) & 1;
        async_wait();
        __syncthreads();
        if (k0 + 32 < K) stage(buf ^ 1, k0 + 32);

        const v16h a0 = load_frag<LDT>(&sA[buf][(wm +  0) * LDT], lane);
        const v16h a1 = load_frag<LDT>(&sA[buf][(wm + 16) * LDT], lane);
        const v16h b0 = load_frag<LDT>(&sB[buf][(wn +  0) * LDT], lane);
        const v16h b1 = load_frag<LDT>(&sB[buf][(wn + 16) * LDT], lane);
        acc00 = wmma_f16(a0, b0, acc00);
        acc01 = wmma_f16(a0, b1, acc01);
        acc10 = wmma_f16(a1, b0, acc10);
        acc11 = wmma_f16(a1, b1, acc11);
    }

    const int nl = lane & 15, mh = (lane >> 4) << 3;
    #pragma unroll
    for (int s = 0; s < 4; ++s) {
        const v8f acc = (s == 0) ? acc00 : (s == 1) ? acc01 : (s == 2) ? acc10 : acc11;
        const int mb = m0 + wm + ((s >> 1) << 4) + mh;
        const int n  = n0 + wn + ((s & 1) << 4) + nl;
        const float bv = bias[n];
        #pragma unroll
        for (int r = 0; r < 8; ++r)
            out[(size_t)(mb + r) * N + n] = (TO)(acc[r] + bv);
    }
}

// ---------------------------------------------------------------------------
// logits[b,h,q,k] = (Qh @ Kh^T) * SCALE, mask==0 -> -inf. HD=64 fits in one
// LDS stage (all async): 128x64 A + 64x64 B + 128x64 mask tile, single
// barrier, then 2 K-chunks of WMMA. Mask read from LDS in the epilogue.
// ---------------------------------------------------------------------------
__global__ __launch_bounds__(256)
void logits_kernel(const _Float16* __restrict__ Qp, const _Float16* __restrict__ Kp,
                   const unsigned char* __restrict__ mask, float* __restrict__ attn)
{
    __shared__ _Float16 sA[128 * LDL];
    __shared__ _Float16 sB[64 * LDL];
    __shared__ unsigned char sM[128 * 64];
    const int tid = threadIdx.x, lane = tid & 31, wave = tid >> 5;
    const int wm = (wave >> 1) << 5, wn = (wave & 1) << 5;
    const int bh = blockIdx.z, b = bh >> 4, h = bh & 15;
    const int q0 = blockIdx.y * 128, kk0 = blockIdx.x * 64;
    const _Float16* Qb = Qp + (size_t)b * NQ_ * DQ_ + h * HD_;
    const _Float16* Kb = Kp + (size_t)b * NK_ * DQ_ + h * HD_;
    const unsigned char* Mb = mask + (size_t)b * NQ_ * NK_ + kk0;

    #pragma unroll
    for (int t = 0; t < 4; ++t) {              // A: 128 q-rows x 64 (full head)
        const int e = (tid + t * 256) << 3;
        const int r = e >> 6, c = e & 63;
        cp_async_b128(&sA[r * LDL + c], Qb + (size_t)(q0 + r) * DQ_ + c);
    }
    #pragma unroll
    for (int t = 0; t < 2; ++t) {              // B: Kp rows are already [n][k]
        const int e = (tid + t * 256) << 3;
        const int n = e >> 6, c = e & 63;
        cp_async_b128(&sB[n * LDL + c], Kb + (size_t)(kk0 + n) * DQ_ + c);
    }
    #pragma unroll
    for (int t = 0; t < 2; ++t) {              // mask tile: 128 x 64 bytes
        const int e = (tid + t * 256) << 4;
        const int r = e >> 6, c = e & 63;
        cp_async_b128(&sM[r * 64 + c], Mb + (size_t)(q0 + r) * NK_ + c);
    }
    async_wait();
    __syncthreads();

    v8f acc00 = {}, acc01 = {}, acc10 = {}, acc11 = {};
    #pragma unroll
    for (int k0 = 0; k0 < HD_; k0 += 32) {
        const v16h a0 = load_frag<LDL>(&sA[(wm +  0) * LDL + k0], lane);
        const v16h a1 = load_frag<LDL>(&sA[(wm + 16) * LDL + k0], lane);
        const v16h b0 = load_frag<LDL>(&sB[(wn +  0) * LDL + k0], lane);
        const v16h b1 = load_frag<LDL>(&sB[(wn + 16) * LDL + k0], lane);
        acc00 = wmma_f16(a0, b0, acc00);
        acc01 = wmma_f16(a0, b1, acc01);
        acc10 = wmma_f16(a1, b0, acc10);
        acc11 = wmma_f16(a1, b1, acc11);
    }

    const float NINF = -__builtin_huge_valf();
    const int nl = lane & 15, mh = (lane >> 4) << 3;
    #pragma unroll
    for (int s = 0; s < 4; ++s) {
        const v8f acc = (s == 0) ? acc00 : (s == 1) ? acc01 : (s == 2) ? acc10 : acc11;
        const int ql = wm + ((s >> 1) << 4) + mh;       // local q row
        const int nn = wn + ((s & 1) << 4) + nl;        // local key col
        #pragma unroll
        for (int r = 0; r < 8; ++r) {
            float v = acc[r] * SCALE_;
            if (!sM[(ql + r) * 64 + nn]) v = NINF;
            attn[((size_t)bh * NQ_ + q0 + ql + r) * NK_ + kk0 + nn] = v;
        }
    }
}

// ---------------------------------------------------------------------------
// Row softmax over the first 256 of each NK=1024 row, in place. Wave-per-row (8 rows per block),
// 8 elems/lane, shfl_xor reductions -- no LDS, no barriers.
// ---------------------------------------------------------------------------
__global__ __launch_bounds__(256)
void softmax_kernel(float* __restrict__ attn)
{
    const int wave = threadIdx.x >> 5, lane = threadIdx.x & 31;
    float* p = attn + ((size_t)blockIdx.x * 8 + wave) * NK_;
    float4 x0 = ((float4*)p)[lane];
    float4 x1 = ((float4*)p)[lane + 32];
    float m = fmaxf(fmaxf(fmaxf(x0.x, x0.y), fmaxf(x0.z, x0.w)),
                    fmaxf(fmaxf(x1.x, x1.y), fmaxf(x1.z, x1.w)));
    #pragma unroll
    for (int d = 16; d > 0; d >>= 1) m = fmaxf(m, __shfl_xor(m, d, 32));
    x0.x = __expf(x0.x - m); x0.y = __expf(x0.y - m);
    x0.z = __expf(x0.z - m); x0.w = __expf(x0.w - m);
    x1.x = __expf(x1.x - m); x1.y = __expf(x1.y - m);
    x1.z = __expf(x1.z - m); x1.w = __expf(x1.w - m);
    float s = (x0.x + x0.y + x0.z + x0.w) + (x1.x + x1.y + x1.z + x1.w);
    #pragma unroll
    for (int d = 16; d > 0; d >>= 1) s += __shfl_xor(s, d, 32);
    const float inv = 1.0f / s;
    x0.x *= inv; x0.y *= inv; x0.z *= inv; x0.w *= inv;
    x1.x *= inv; x1.y *= inv; x1.z *= inv; x1.w *= inv;
    ((float4*)p)[lane] = x0;
    ((float4*)p)[lane + 32] = x1;
}

// ---------------------------------------------------------------------------
// ctx[b,q, 64h+d] = attn[b,h,q,:] @ Vh[:,d]. Block tile: 128 q x 64 hd.
// A tile is fp32->f16 convert, B tile transposed gather (no async path).
// ---------------------------------------------------------------------------
__global__ __launch_bounds__(256)
void ctx_kernel(const float* __restrict__ attn, const _Float16* __restrict__ Vp,
                _Float16* __restrict__ ctx)
{
    __shared__ _Float16 sA[2][128 * LDT];
    __shared__ _Float16 sB[2][64 * LDT];
    const int tid = threadIdx.x, lane = tid & 31, wave = tid >> 5;
    const int wm = (wave >> 1) << 5, wn = (wave & 1) << 5;
    const int bh = blockIdx.y, b = bh >> 4, h = bh & 15;
    const int q0 = blockIdx.x * 128;
    const float*    Ab = attn + (size_t)bh * NQ_ * NK_;
    const _Float16* Vb = Vp + (size_t)b * NK_ * DQ_ + h * HD_;

    auto stage = [&](int buf, int k0) {
        #pragma unroll
        for (int t = 0; t < 4; ++t) {          // A: attn fp32 -> f16
            const int e = (tid + t * 256) << 2;
            const int r = e >> 5, c = e & 31;
            const float4 v = *(const float4*)(Ab + (size_t)(q0 + r) * NK_ + k0 + c);
            v4h hh = { (_Float16)v.x, (_Float16)v.y, (_Float16)v.z, (_Float16)v.w };
            *(v4h*)(&sA[buf][r * LDT + c]) = hh;
        }
        {                                      // B: sB[d][k] = Vh[k][d] (transpose)
            const int e  = tid << 3;
            const int kk = e >> 6, n = e & 63;
            const v8h v = *(const v8h*)(Vb + (size_t)(k0 + kk) * DQ_ + n);
            #pragma unroll
            for (int j = 0; j < 8; ++j) sB[buf][(n + j) * LDT + kk] = v[j];
        }
    };

    v8f acc00 = {}, acc01 = {}, acc10 = {}, acc11 = {};
    stage(0, 0);
    for (int k0 = 0; k0 < NK_; k0 += 32) {
        const int buf = (k0 >> 5) & 1;
        __syncthreads();
        if (k0 + 32 < NK_) stage(buf ^ 1, k0 + 32);

        const v16h a0 = load_frag<LDT>(&sA[buf][(wm +  0) * LDT], lane);
        const v16h a1 = load_frag<LDT>(&sA[buf][(wm + 16) * LDT], lane);
        const v16h b0 = load_frag<LDT>(&sB[buf][(wn +  0) * LDT], lane);
        const v16h b1 = load_frag<LDT>(&sB[buf][(wn + 16) * LDT], lane);
        acc00 = wmma_f16(a0, b0, acc00);
        acc01 = wmma_f16(a0, b1, acc01);
        acc10 = wmma_f16(a1, b0, acc10);
        acc11 = wmma_f16(a1, b1, acc11);
    }

    const int nl = lane & 15, mh = (lane >> 4) << 3;
    #pragma unroll
    for (int s = 0; s < 4; ++s) {
        const v8f acc = (s == 0) ? acc00 : (s == 1) ? acc01 : (s == 2) ? acc10 : acc11;
        const int qb = q0 + wm + ((s >> 1) << 4) + mh;
        const int n  = wn + ((s & 1) << 4) + nl;
        #pragma unroll
        for (int r = 0; r < 8; ++r)
            ctx[((size_t)b * NQ_ + qb + r) * DQ_ + h * HD_ + n] = (_Float16)acc[r];
    }
}

// ---------------------------------------------------------------------------
extern "C" void kernel_launch(void* const* d_in, const int* in_sizes, int n_in,
                              void* d_out, int out_size, void* d_ws, size_t ws_size,
                              hipStream_t stream)
{
    const float* q    = (const float*)d_in[0];
    const float* k    = (const float*)d_in[1];
    const float* v    = (const float*)d_in[2];
    const unsigned char* mask = (const unsigned char*)d_in[3];
    const float* Wq = (const float*)d_in[4];  const float* bq = (const float*)d_in[5];
    const float* Wk = (const float*)d_in[6];  const float* bk = (const float*)d_in[7];
    const float* Wv = (const float*)d_in[8];  const float* bv = (const float*)d_in[9];
    const float* Wo = (const float*)d_in[10]; const float* bo = (const float*)d_in[11];

    float* out  = (float*)d_out;
    float* attn = out + (size_t)B_SZ * NQ_ * DQ_;          // second tuple output

    const size_t PROJ = (size_t)B_SZ * NQ_ * DQ_;          // 8M f16 elems each
    _Float16* Qp  = (_Float16*)d_ws;
    _Float16* Kp  = Qp + PROJ;
    _Float16* Vp  = Kp + PROJ;
    _Float16* ctx = Vp + PROJ;

    const dim3 blk(256);
    const int M = B_SZ * NQ_;                               // 8192 rows

    gemm_bias_kernel<float, _Float16><<<dim3(16, 64), blk, 0, stream>>>(q, Wq, bq, Qp, M, DQ_, 1024);
    gemm_bias_kernel<float, _Float16><<<dim3(16, 64), blk, 0, stream>>>(k, Wk, bk, Kp, M, DQ_, 768);
    gemm_bias_kernel<float, _Float16><<<dim3(16, 64), blk, 0, stream>>>(v, Wv, bv, Vp, M, DQ_, 768);
    logits_kernel<<<dim3(NK_ / 64, NQ_ / 128, B_SZ * H_), blk, 0, stream>>>(Qp, Kp, mask, attn);
    softmax_kernel<<<dim3(B_SZ * H_ * NQ_ / 8), blk, 0, stream>>>(attn);
    ctx_kernel<<<dim3(NQ_ / 128, B_SZ * H_), blk, 0, stream>>>(attn, Vp, ctx);
    gemm_bias_kernel<_Float16, float><<<dim3(16, 64), blk, 0, stream>>>(ctx, Wo, bo, out, M, DQ_, 1024);
}
